// SpatioTemporalAttention_79706003079269
// MI455X (gfx1250) — matrix operand order repaired and checked by reading, not compiled, not run on hardware
//
#include <hip/hip_runtime.h>
#include <hip/hip_bf16.h>

// Problem constants (from reference)
#define Bb 2
#define Ll 8
#define Nn 256
#define Cc 1024
#define Hh 16
#define Dd 64
#define Tt 2048          // L*N
#define CONDc 64
#define BT (Bb*Tt)       // 4096
#define KPANEL 128       // K-panel staged in LDS by the TDM

typedef __attribute__((ext_vector_type(16))) _Float16 v16h;
typedef __attribute__((ext_vector_type(8)))  _Float16 v8h;
typedef __attribute__((ext_vector_type(8)))  float    v8f;
typedef __attribute__((ext_vector_type(4)))  int      i32x4;
typedef __attribute__((ext_vector_type(8)))  int      i32x8;

__device__ __forceinline__ v8f wmma_f16(v16h a, v16h b, v8f c) {
    return __builtin_amdgcn_wmma_f32_16x16x32_f16(false, a, false, b, (short)0, c, false, false);
}

// ---------------------------------------------------------------------------
// ss = cond @ W_ada + b_ada        (B x 2C, tiny)
// ---------------------------------------------------------------------------
__global__ __launch_bounds__(256) void cond_proj_kernel(
    const float* __restrict__ cond, const float* __restrict__ Wada,
    const float* __restrict__ bada, float* __restrict__ ss) {
    int j = blockIdx.x * blockDim.x + threadIdx.x;   // 0 .. B*2C-1
    int b = j >> 11;                                  // 2C = 2048
    int o = j & 2047;
    float acc = bada[o];
    #pragma unroll 8
    for (int i = 0; i < CONDc; ++i)
        acc += cond[b * CONDc + i] * Wada[(size_t)i * 2048 + o];
    ss[j] = acc;
}

// ---------------------------------------------------------------------------
// AdaLN: layernorm + (1+scale)*x + shift  -> f16   (wave per token)
// ---------------------------------------------------------------------------
__global__ __launch_bounds__(256) void adaln_kernel(
    const float* __restrict__ x, const float* __restrict__ ss,
    _Float16* __restrict__ xh) {
    int lane  = threadIdx.x & 31;
    int wv    = threadIdx.x >> 5;
    int token = blockIdx.x * 8 + wv;       // B*T tokens, exact
    int b     = token >> 11;               // T = 2048
    const float* xp = x + (size_t)token * Cc;

    float vals[32];
    float sum = 0.f, sq = 0.f;
    #pragma unroll
    for (int ch = 0; ch < 8; ++ch) {
        float4 v4 = *(const float4*)(xp + ch * 128 + lane * 4);
        vals[ch*4+0] = v4.x; vals[ch*4+1] = v4.y; vals[ch*4+2] = v4.z; vals[ch*4+3] = v4.w;
        sum += v4.x + v4.y + v4.z + v4.w;
        sq  += v4.x*v4.x + v4.y*v4.y + v4.z*v4.z + v4.w*v4.w;
    }
    #pragma unroll
    for (int off = 16; off >= 1; off >>= 1) {
        sum += __shfl_xor(sum, off, 32);
        sq  += __shfl_xor(sq,  off, 32);
    }
    float mean = sum * (1.f / 1024.f);
    float var  = sq  * (1.f / 1024.f) - mean * mean;
    float rstd = rsqrtf(var + 1e-5f);

    const float* shift = ss + (size_t)b * 2048;
    const float* scl   = shift + 1024;
    _Float16* op = xh + (size_t)token * Cc;
    #pragma unroll
    for (int ch = 0; ch < 8; ++ch)
        #pragma unroll
        for (int q = 0; q < 4; ++q) {
            int c = ch * 128 + lane * 4 + q;
            float xn = (vals[ch*4+q] - mean) * rstd;
            op[c] = (_Float16)(xn * (1.f + scl[c]) + shift[c]);
        }
}

// ---------------------------------------------------------------------------
// Convert+transpose fp32 weight [K][N] -> f16 [N][K]
// ---------------------------------------------------------------------------
__global__ __launch_bounds__(256) void wconv_kernel(
    const float* __restrict__ W, _Float16* __restrict__ Wt) {
    int idx = blockIdx.x * blockDim.x + threadIdx.x;  // n*K + k
    int n = idx >> 10;
    int k = idx & 1023;
    Wt[idx] = (_Float16)W[(size_t)k * Cc + n];
}

// ---------------------------------------------------------------------------
// WMMA GEMM: Out[M][N] = A[M][K](f16) @ Bt[N][K]^T(f16) + bias[N]
//   * block = 128 threads (4 waves) computing a 128M x 64N C-block
//   * wave  = 32M x 64N  (2 A-fragments x 4 B-fragments -> 8 WMMAs / K-step)
//   * B panel (64 x KPANEL f16 = 16 KB) staged in LDS by the Tensor Data
//     Mover: one wave issues tensor_load_to_lds with a 2D D#, waits on
//     TENSORcnt, block barrier publishes the panel.
// ---------------------------------------------------------------------------
template <typename OutT>
__global__ __launch_bounds__(128) void gemm_kernel(
    const _Float16* __restrict__ A, const _Float16* __restrict__ Bt,
    const float* __restrict__ bias, OutT* __restrict__ Out,
    int M, int N, int K) {
    __shared__ __align__(32) _Float16 bpanel[64][KPANEL];

    int lane  = threadIdx.x & 31;
    int wv    = threadIdx.x >> 5;
    int nblk  = blockIdx.x & ((N >> 6) - 1);   // N/64 blocks
    int mblk  = blockIdx.x / (N >> 6);
    int nbase = nblk * 64;
    int mbase = mblk * 128 + wv * 32;
    int ln    = lane & 15;
    int l16   = (lane >> 4) & 1;
    int abase = l16 * 16;

    const _Float16* arow0 = A + (size_t)(mbase + ln) * K;
    const _Float16* arow1 = A + (size_t)(mbase + 16 + ln) * K;

    v8f acc[2][4] = {};

    unsigned int lds_base = (unsigned int)(size_t)(&bpanel[0][0]);

    for (int kp = 0; kp < K; kp += KPANEL) {
        __syncthreads();                        // previous panel fully consumed
        if (wv == 0) {
            // --- Tensor DMA descriptor (D#), groups 0 and 1 (2D tensor) ---
            unsigned long long ga =
                (unsigned long long)(size_t)(Bt + (size_t)nbase * K + kp);
            i32x4 g0;
            g0[0] = 1;                                   // count=1, user desc
            g0[1] = (int)lds_base;                       // lds_addr
            g0[2] = (int)(unsigned int)(ga & 0xffffffffull);
            g0[3] = (int)((unsigned int)((ga >> 32) & 0x01ffffffull)
                          | 0x80000000u);                // type=2 ("image")
            i32x8 g1;
            g1[0] = 0x00010000;        // workgroup_mask=0, data_size=1 (2B)
            g1[1] = 0x04000000;        // tensor_dim0 = 1024 (bits 79:48)
            g1[2] = 0x04000000;        // tensor_dim1 = 1024 (bits 111:80)
            g1[3] = (KPANEL << 16);    // tile_dim0 = 128  (bits 127:112)
            g1[4] = 64;                // tile_dim1 = 64   (bits 143:128)
            g1[5] = 1024;              // tensor_dim0_stride = K (bits 207:160)
            g1[6] = 0;
            g1[7] = 0;
            asm volatile("tensor_load_to_lds %0, %1" :: "s"(g0), "s"(g1) : "memory");
            __builtin_amdgcn_s_wait_tensorcnt(0);
        }
        __syncthreads();                        // panel visible to all waves

        #pragma unroll
        for (int k1 = 0; k1 < KPANEL; k1 += 32) {
            int k0 = kp + k1;
            __builtin_prefetch(arow0 + k0 + 256, 0, 1);
            __builtin_prefetch(arow1 + k0 + 256, 0, 1);
            // A fragments (contiguous K layout: lanes 0-15 K0-15, 16-31 K16-31)
            v8h a00 = *(const v8h*)(arow0 + k0 + abase);
            v8h a01 = *(const v8h*)(arow0 + k0 + abase + 8);
            v8h a10 = *(const v8h*)(arow1 + k0 + abase);
            v8h a11 = *(const v8h*)(arow1 + k0 + abase + 8);
            v16h a0, a1;
            #pragma unroll
            for (int j = 0; j < 8; ++j) {
                a0[j] = a00[j]; a0[j + 8] = a01[j];
                a1[j] = a10[j]; a1[j + 8] = a11[j];
            }
            #pragma unroll
            for (int j = 0; j < 4; ++j) {
                const _Float16* bl = &bpanel[j * 16 + ln][k1 + l16 * 16];
                v16h bfrag = *(const v16h*)bl;
                acc[0][j] = wmma_f16(a0, bfrag, acc[0][j]);
                acc[1][j] = wmma_f16(a1, bfrag, acc[1][j]);
            }
        }
    }

    // epilogue: bias add + store
    #pragma unroll
    for (int j = 0; j < 4; ++j) {
        int col = nbase + j * 16 + ln;
        float bsc = bias[col];
        #pragma unroll
        for (int mi = 0; mi < 2; ++mi)
            #pragma unroll
            for (int r = 0; r < 8; ++r) {
                int row = mbase + mi * 16 + r + l16 * 8;
                Out[(size_t)row * N + col] = (OutT)(acc[mi][j][r] + bsc);
            }
    }
}

// ---------------------------------------------------------------------------
// 2D RoPE + repack [B*T, C] f16 -> [B,H,T,D] f16   (thread per (b,t,h,i<16))
// ---------------------------------------------------------------------------
__global__ __launch_bounds__(256) void rope_kernel(
    const _Float16* __restrict__ src, const int* __restrict__ seq_idx,
    const int* __restrict__ frame_idx, _Float16* __restrict__ dst) {
    int tid = blockIdx.x * blockDim.x + threadIdx.x; // ((b*T+t)*H+h)*16+i
    int i = tid & 15;
    int h = (tid >> 4) & 15;
    int t = (tid >> 8) & 2047;
    int b = tid >> 19;
    float pos_s = (float)seq_idx[b * Nn + (t & 255)];
    float pos_f = (float)frame_idx[t >> 8];
    // inv_freq = 10000^(-i/16) = exp(-i * ln(10000)/16)
    float inv = __expf(-(float)i * 0.5756462732485114f);
    float as = pos_s * inv, af = pos_f * inv;
    float cs = __cosf(as), sn = __sinf(as);
    float cf = __cosf(af), sf = __sinf(af);
    const _Float16* sp = src + ((size_t)(b * Tt + t)) * Cc + h * Dd;
    float x1 = (float)sp[i],      x2 = (float)sp[16 + i];
    float y1 = (float)sp[32 + i], y2 = (float)sp[48 + i];
    _Float16* dp = dst + (((size_t)(b * Hh + h)) * Tt + t) * Dd;
    dp[i]      = (_Float16)(x1 * cs - x2 * sn);
    dp[16 + i] = (_Float16)(x1 * sn + x2 * cs);
    dp[32 + i] = (_Float16)(y1 * cf - y2 * sf);
    dp[48 + i] = (_Float16)(y1 * sf + y2 * cf);
}

// ---------------------------------------------------------------------------
// Repack V: [B*T, C] f16 -> transposed [B,H,D,T] f16
// ---------------------------------------------------------------------------
__global__ __launch_bounds__(256) void vpack_kernel(
    const _Float16* __restrict__ v, _Float16* __restrict__ vt) {
    int idx = blockIdx.x * blockDim.x + threadIdx.x;  // over B*T*C
    int c  = idx & 1023;
    int bt = idx >> 10;
    int b  = bt >> 11;
    int t  = bt & 2047;
    int h  = c >> 6;
    int d  = c & 63;
    vt[(((size_t)(b * Hh + h)) * Dd + d) * Tt + t] = v[idx];
}

// ---------------------------------------------------------------------------
// Flash attention: one wave per (b, h, 16-query tile).
// Frame-causality handled by the key loop bound (16 | 256), padding by bias.
// ---------------------------------------------------------------------------
__global__ __launch_bounds__(128) void attn_kernel(
    const _Float16* __restrict__ Qh,   // [B,H,T,D]
    const _Float16* __restrict__ Kh,   // [B,H,T,D]
    const _Float16* __restrict__ Vt,   // [B,H,D,T]
    const int* __restrict__ mask,      // [B,T]
    _Float16* __restrict__ Oh) {       // [B*T, C]  (c = h*D+d)
    __shared__ __align__(32) _Float16 pbuf[4][16][32];

    int lane = threadIdx.x & 31;
    int wv   = threadIdx.x >> 5;
    int wid  = blockIdx.x * 4 + wv;               // B*H*(T/16) tiles, exact
    int b    = wid / (Hh * (Tt / 16));
    int rem  = wid % (Hh * (Tt / 16));
    int h    = rem / (Tt / 16);
    int qt   = rem % (Tt / 16);
    int qbase = qt * 16;
    int fq    = qbase >> 8;                        // frame of this query tile
    int smax  = (fq + 1) * Nn;                     // causal bound (frame gran.)
    int ln  = lane & 15;
    int l16 = (lane >> 4) & 1;
    int abase = l16 * 16;

    // Q fragments (two K-chunks of the D=64 reduction)
    const _Float16* qp = Qh + ((size_t)(b * Hh + h) * Tt + qbase + ln) * Dd;
    v16h aq0, aq1;
    {
        v8h q00 = *(const v8h*)(qp + abase);
        v8h q01 = *(const v8h*)(qp + abase + 8);
        v8h q10 = *(const v8h*)(qp + 32 + abase);
        v8h q11 = *(const v8h*)(qp + 32 + abase + 8);
        #pragma unroll
        for (int j = 0; j < 8; ++j) {
            aq0[j] = q00[j]; aq0[j + 8] = q01[j];
            aq1[j] = q10[j]; aq1[j + 8] = q11[j];
        }
    }

    v8f acc0 = {}, acc1 = {}, acc2 = {}, acc3 = {};
    float mstat[8], lstat[8];
    #pragma unroll
    for (int r = 0; r < 8; ++r) { mstat[r] = -1e30f; lstat[r] = 0.f; }

    const _Float16* kbase = Kh + (size_t)(b * Hh + h) * Tt * Dd;
    const _Float16* vbase = Vt + (size_t)(b * Hh + h) * Dd * Tt;
    const int* mrow = mask + b * Tt;

    for (int s0 = 0; s0 < smax; s0 += 32) {
        // --- logits for two 16-key subtiles (QK^T, K-dim = D = 64) ---
        v8f sA = {}, sB = {};
        {
            const _Float16* kp = kbase + (size_t)(s0 + ln) * Dd + l16 * 16;
            v16h kb0 = *(const v16h*)(kp);
            v16h kb1 = *(const v16h*)(kp + 32);
            sA = wmma_f16(aq0, kb0, sA);
            sA = wmma_f16(aq1, kb1, sA);
            kp += 16 * Dd;
            kb0 = *(const v16h*)(kp);
            kb1 = *(const v16h*)(kp + 32);
            sB = wmma_f16(aq0, kb0, sB);
            sB = wmma_f16(aq1, kb1, sB);
        }
        float bias0 = (mrow[s0 + ln] == 0)      ? -1e30f : 0.f;
        float bias1 = (mrow[s0 + 16 + ln] == 0) ? -1e30f : 0.f;

        // --- online softmax (row stats replicated across each 16-lane half) ---
        float p0[8], p1[8];
        #pragma unroll
        for (int r = 0; r < 8; ++r) {
            float sa = sA[r] * 0.125f + bias0;   // scale = 1/sqrt(64)
            float sb = sB[r] * 0.125f + bias1;
            float mx = fmaxf(sa, sb);
            #pragma unroll
            for (int off = 8; off >= 1; off >>= 1)
                mx = fmaxf(mx, __shfl_xor(mx, off, 16));
            float mn = fmaxf(mstat[r], mx);
            float alpha = __expf(mstat[r] - mn);
            float e0 = __expf(sa - mn);
            float e1 = __expf(sb - mn);
            float rs = e0 + e1;
            #pragma unroll
            for (int off = 8; off >= 1; off >>= 1)
                rs += __shfl_xor(rs, off, 16);
            lstat[r] = lstat[r] * alpha + rs;
            mstat[r] = mn;
            acc0[r] *= alpha; acc1[r] *= alpha; acc2[r] *= alpha; acc3[r] *= alpha;
            p0[r] = e0; p1[r] = e1;
        }

        // --- stage P (C/D layout) through LDS, reload in A-fragment layout ---
        int prow = l16 * 8;
        #pragma unroll
        for (int r = 0; r < 8; ++r) {
            pbuf[wv][prow + r][ln]      = (_Float16)p0[r];
            pbuf[wv][prow + r][ln + 16] = (_Float16)p1[r];
        }
        asm volatile("s_wait_dscnt 0" ::: "memory");
        v16h pa;
        {
            const _Float16* pr = &pbuf[wv][ln][0];
            v8h pA = *(const v8h*)(pr + abase);
            v8h pB = *(const v8h*)(pr + abase + 8);
            #pragma unroll
            for (int j = 0; j < 8; ++j) { pa[j] = pA[j]; pa[j + 8] = pB[j]; }
        }

        // --- PV: 4 WMMAs covering D = 64 ---
        const _Float16* vp = vbase + (size_t)ln * Tt + s0 + l16 * 16;
        acc0 = wmma_f16(pa, *(const v16h*)(vp),            acc0);
        acc1 = wmma_f16(pa, *(const v16h*)(vp + 16 * Tt),  acc1);
        acc2 = wmma_f16(pa, *(const v16h*)(vp + 32 * Tt),  acc2);
        acc3 = wmma_f16(pa, *(const v16h*)(vp + 48 * Tt),  acc3);
    }

    // --- epilogue: normalize and scatter back to [B*T, C] layout ---
    #pragma unroll
    for (int r = 0; r < 8; ++r) {
        float inv = 1.f / lstat[r];
        int t = qbase + r + l16 * 8;
        size_t orow = ((size_t)b * Tt + t) * Cc + h * Dd;
        Oh[orow + ln]      = (_Float16)(acc0[r] * inv);
        Oh[orow + 16 + ln] = (_Float16)(acc1[r] * inv);
        Oh[orow + 32 + ln] = (_Float16)(acc2[r] * inv);
        Oh[orow + 48 + ln] = (_Float16)(acc3[r] * inv);
    }
}

// ---------------------------------------------------------------------------
extern "C" void kernel_launch(void* const* d_in, const int* in_sizes, int n_in,
                              void* d_out, int out_size, void* d_ws, size_t ws_size,
                              hipStream_t stream) {
    const float* s_frames  = (const float*)d_in[0];
    const int*   frame_idx = (const int*)  d_in[1];
    const int*   seq_idx   = (const int*)  d_in[2];
    const int*   mask      = (const int*)  d_in[3];
    const float* cond      = (const float*)d_in[4];
    const float* W_ada     = (const float*)d_in[5];
    const float* b_ada     = (const float*)d_in[6];
    const float* Wq = (const float*)d_in[7];  const float* bq = (const float*)d_in[8];
    const float* Wk = (const float*)d_in[9];  const float* bk = (const float*)d_in[10];
    const float* Wv = (const float*)d_in[11]; const float* bv = (const float*)d_in[12];
    const float* Wo = (const float*)d_in[13]; const float* bo = (const float*)d_in[14];
    float* out = (float*)d_out;

    // workspace carve-out
    char* p = (char*)d_ws;
    auto alloc = [&](size_t bytes) -> void* {
        void* r = (void*)p;
        p += (bytes + 255) & ~(size_t)255;
        return r;
    };
    float*     ss  = (float*)    alloc(Bb * 2 * Cc * sizeof(float));
    _Float16*  xh  = (_Float16*) alloc((size_t)BT * Cc * 2);
    _Float16*  wqt = (_Float16*) alloc((size_t)Cc * Cc * 2);
    _Float16*  wkt = (_Float16*) alloc((size_t)Cc * Cc * 2);
    _Float16*  wvt = (_Float16*) alloc((size_t)Cc * Cc * 2);
    _Float16*  wot = (_Float16*) alloc((size_t)Cc * Cc * 2);
    _Float16*  qf  = (_Float16*) alloc((size_t)BT * Cc * 2);
    _Float16*  kf  = (_Float16*) alloc((size_t)BT * Cc * 2);
    _Float16*  vf  = (_Float16*) alloc((size_t)BT * Cc * 2);
    _Float16*  qr  = (_Float16*) alloc((size_t)BT * Cc * 2);
    _Float16*  kr  = (_Float16*) alloc((size_t)BT * Cc * 2);
    _Float16*  vt  = (_Float16*) alloc((size_t)BT * Cc * 2);
    _Float16*  xo  = (_Float16*) alloc((size_t)BT * Cc * 2);
    (void)ws_size; (void)in_sizes; (void)n_in; (void)out_size;

    // 1) weight convert+transpose to f16 [N][K]
    wconv_kernel<<<dim3((Cc * Cc) / 256), dim3(256), 0, stream>>>(Wq, wqt);
    wconv_kernel<<<dim3((Cc * Cc) / 256), dim3(256), 0, stream>>>(Wk, wkt);
    wconv_kernel<<<dim3((Cc * Cc) / 256), dim3(256), 0, stream>>>(Wv, wvt);
    wconv_kernel<<<dim3((Cc * Cc) / 256), dim3(256), 0, stream>>>(Wo, wot);

    // 2) adaln modulation params + layernorm
    cond_proj_kernel<<<dim3((Bb * 2 * Cc) / 256), dim3(256), 0, stream>>>(
        cond, W_ada, b_ada, ss);
    adaln_kernel<<<dim3(BT / 8), dim3(256), 0, stream>>>(s_frames, ss, xh);

    // 3) Q/K/V projections (WMMA GEMMs, TDM-staged B panels)
    int gemm_blocks = (BT / 128) * (Cc / 64);   // 32 * 16 = 512
    gemm_kernel<_Float16><<<dim3(gemm_blocks), dim3(128), 0, stream>>>(
        xh, wqt, bq, qf, BT, Cc, Cc);
    gemm_kernel<_Float16><<<dim3(gemm_blocks), dim3(128), 0, stream>>>(
        xh, wkt, bk, kf, BT, Cc, Cc);
    gemm_kernel<_Float16><<<dim3(gemm_blocks), dim3(128), 0, stream>>>(
        xh, wvt, bv, vf, BT, Cc, Cc);

    // 4) RoPE + head-major repack; V transposed repack
    rope_kernel<<<dim3((BT * Hh * 16) / 256), dim3(256), 0, stream>>>(
        qf, seq_idx, frame_idx, qr);
    rope_kernel<<<dim3((BT * Hh * 16) / 256), dim3(256), 0, stream>>>(
        kf, seq_idx, frame_idx, kr);
    vpack_kernel<<<dim3((BT * Cc) / 256), dim3(256), 0, stream>>>(vf, vt);

    // 5) flash attention (WMMA)
    attn_kernel<<<dim3(Bb * Hh * (Tt / 16) / 4), dim3(128), 0, stream>>>(
        qr, kr, vt, mask, xo);

    // 6) output projection (WMMA, f32 out)
    gemm_kernel<float><<<dim3(gemm_blocks), dim3(128), 0, stream>>>(
        xo, wot, bo, out, BT, Cc, Cc);
}
